// ECGTransformer_23278722744933
// MI455X (gfx1250) — matrix operand order repaired and checked by reading, not compiled, not run on hardware
//
#include <hip/hip_runtime.h>
#include <hip/hip_bf16.h>

// ---------------------------------------------------------------------------
// ECG Transformer, fully fused per batch element, CDNA5 WMMA f16 path.
// ---------------------------------------------------------------------------

typedef __attribute__((ext_vector_type(16))) _Float16 v16h;
typedef __attribute__((ext_vector_type(8)))  float    v8f;

#define NBATCH 4096
#define TOK    64          // tokens per sequence
#define DM     64          // model dim
#define NHEADS 4
#define HD     16
#define FFDIM  256
#define NLAYER 6
#define XP     68          // f32 LDS pitch (64 + 4 pad -> bank spread)
#define HP     264         // f16 LDS pitch for FF hidden (256 + 8 pad)

// ---- packed-weight layout (halves) ----------------------------------------
#define TILE_H  512                              // one 32x16 B tile (32 lanes * 16 halves)
#define SZ_W16  512                              // 16x16 padded to 32x16
#define SZ_FCW  (8 * 512)                        // 64x64  -> 2x4 tiles
#define SZ_W1   (32 * 512)                       // 64x256 -> 2x16 tiles
#define SZ_W2   (32 * 512)                       // 256x64 -> 8x4 tiles
#define SZ_TB   (3 * SZ_W16 + SZ_FCW + SZ_W1 + SZ_W2)   // 38400
#define SZ_SA   (3 * SZ_W16 + SZ_FCW)                   // 5632

#define OFF_WV  0
#define OFF_WK  SZ_W16
#define OFF_WQ  (2 * SZ_W16)
#define OFF_FCW (3 * SZ_W16)
#define OFF_W1  (3 * SZ_W16 + SZ_FCW)
#define OFF_W2  (3 * SZ_W16 + SZ_FCW + SZ_W1)

#define ENC_TB_OFF(e) ((e) * SZ_TB)
#define DEC_SA_OFF(d) (6 * SZ_TB + (d) * SZ_SA)
#define DEC_TB_OFF(d) (6 * SZ_TB + 6 * SZ_SA + (d) * SZ_TB)
#define PACK_TOTAL_HALVES (6 * SZ_TB + 6 * SZ_SA + 6 * SZ_TB)

// ---- flattened input indexing (setup_inputs dict order) --------------------
// 0 src, 1 trg, 2 sampling_w, 3 init_bl_w, 4 enc_pos, 5 dec_pos,
// enc layer e (13): wv wk wq fcw fcb ln1w ln1b ln2w ln2b w1 b1 w2 b2
// dec layer d (20): wv wk wq fcw fcb lnw lnb | blk: wv wk wq fcw fcb ln1w ln1b ln2w ln2b w1 b1 w2 b2
#define IDX_SRC    0
#define IDX_TRG    1
#define IDX_SAMP   2
#define IDX_INITBL 3
#define IDX_EPOS   4
#define IDX_DPOS   5
#define ENC_BASE(e) (6 + (e) * 13)
#define DEC_BASE(d) (84 + (d) * 20)
#define N_INPUTS   204

struct PtrTable { const float* p[N_INPUTS]; };

// ---------------------------------------------------------------------------
// WMMA + fragment helpers (ISA 7.12.2 layouts, wave32)
// ---------------------------------------------------------------------------
__device__ __forceinline__ v8f wmma_f16(v16h a, v16h b, v8f c) {
  return __builtin_amdgcn_wmma_f32_16x16x32_f16(false, a, false, b, (short)0, c,
                                                false, false);
}

// A fragment 16x32 from f32 LDS (row-major, pitch XP).
// lane l: m = row0+(l&15); kb = (l>=16)?8:0; halves 0..7 -> K=kb+i, 8..15 -> K=16+kb+i
__device__ __forceinline__ v16h loadA32(const float* X, int row0, int col0, int lane) {
  int m  = row0 + (lane & 15);
  int kb = (lane >> 4) * 8;
  const float* p0 = X + m * XP + col0 + kb;
  v16h a;
#pragma unroll
  for (int i = 0; i < 8; ++i) a[i] = (_Float16)p0[i];
#pragma unroll
  for (int i = 0; i < 8; ++i) a[8 + i] = (_Float16)p0[16 + i];
  return a;
}

// A fragment with only K=0..15 valid (head-dim GEMMs); K=16..31 zero.
__device__ __forceinline__ v16h loadA32_pad16(const float* X, int row0, int col0, int lane) {
  int m  = row0 + (lane & 15);
  int kb = (lane >> 4) * 8;
  const float* p0 = X + m * XP + col0 + kb;
  v16h a;
#pragma unroll
  for (int i = 0; i < 8; ++i) a[i] = (_Float16)p0[i];
#pragma unroll
  for (int i = 0; i < 8; ++i) a[8 + i] = (_Float16)0.f;
  return a;
}

// A fragment from f16 LDS (FF hidden, pitch HP).
__device__ __forceinline__ v16h loadAh(const _Float16* H, int row0, int col0, int lane) {
  int m  = row0 + (lane & 15);
  int kb = (lane >> 4) * 8;
  const _Float16* p0 = H + m * HP + col0 + kb;
  v16h a;
#pragma unroll
  for (int i = 0; i < 8; ++i) a[i] = p0[i];
#pragma unroll
  for (int i = 0; i < 8; ++i) a[8 + i] = p0[16 + i];
  return a;
}

// B fragment from pre-packed global f16 weights: lane l reads 16 contiguous halves.
__device__ __forceinline__ v16h loadBpack(const _Float16* W, int lane) {
  return *(const v16h*)(W + lane * 16);
}

// B = K_h^T (scores GEMM): B[k][n] = Ksrc[token n0+(l&15)][hbase+k]; rows 16..31 zero.
__device__ __forceinline__ v16h loadB_kT(const float* Ksrc, int n0, int hbase, int lane) {
  v16h b;
#pragma unroll
  for (int i = 0; i < 16; ++i) b[i] = (_Float16)0.f;
  {
    const float* p = Ksrc + (n0 + (lane & 15)) * XP + hbase + (lane >> 4) * 8;
#pragma unroll
    for (int i = 0; i < 8; ++i) b[i] = (_Float16)p[i];
  }
  return b;
}

// B = V_h (PV GEMM): B[k][n] = V[token k0+kb+(i<8?i:i+8)][hbase + (l&15)], kb=8*(l>=16), strided column read.
__device__ __forceinline__ v16h loadB_V(const float* V, int k0, int hbase, int lane) {
  int nn = lane & 15;
  int kb = (lane >> 4) * 8;
  const float* p = V + (k0 + kb) * XP + hbase + nn;
  v16h b;
#pragma unroll
  for (int i = 0; i < 16; ++i) b[i] = (_Float16)p[(i < 8 ? i : i + 8) * XP];
  return b;
}

// C/D tile store to f32 LDS: VGPR r -> M = row0 + (l>=16?8:0) + r, N = col0 + (l&15)
__device__ __forceinline__ void storeC(float* D, int row0, int col0, int lane, v8f c) {
  int nn = col0 + (lane & 15);
  int mb = row0 + (lane >> 4) * 8;
#pragma unroll
  for (int r = 0; r < 8; ++r) D[(mb + r) * XP + nn] = c[r];
}

// store C to f16 LDS with bias + relu (FF hidden)
__device__ __forceinline__ void storeH(_Float16* H, int row0, int col0, int lane, v8f c,
                                       const float* b1) {
  int nn = col0 + (lane & 15);
  int mb = row0 + (lane >> 4) * 8;
  float bias = b1[nn];
#pragma unroll
  for (int r = 0; r < 8; ++r) {
    float v = c[r] + bias;
    v = v > 0.f ? v : 0.f;
    H[(mb + r) * HP + nn] = (_Float16)v;
  }
}

// LayerNorm of (sa + sb) -> dst, per row; two-pass to avoid register arrays.
__device__ __forceinline__ void ln_resid(float* dst, const float* sa, const float* sb,
                                         const float* w, const float* b, int tid) {
  if (tid < TOK) {
    const float* ra = sa + tid * XP;
    const float* rb = sb + tid * XP;
    float sum = 0.f, sq = 0.f;
#pragma unroll 8
    for (int c = 0; c < DM; ++c) { float x = ra[c] + rb[c]; sum += x; sq += x * x; }
    float m    = sum * (1.f / DM);
    float var  = sq * (1.f / DM) - m * m;
    float rinv = rsqrtf(var + 1e-5f);
    float* rd = dst + tid * XP;
#pragma unroll 8
    for (int c = 0; c < DM; ++c) {
      float x = ra[c] + rb[c];
      rd[c] = (x - m) * rinv * w[c] + b[c];
    }
  }
}

// ---------------------------------------------------------------------------
// attention + residual + layernorm:  Xdst = LN( fc(MHA(VKsrc, VKsrc, Qsrc)) + Qsrc )
// ---------------------------------------------------------------------------
__device__ void attn_ln(float* Xdst, const float* VKsrc, const float* Qsrc,
                        const _Float16* Wt, const float* fcb,
                        const float* lnw, const float* lnb,
                        float* Qb, float* Kb, float* Vb, float* Sb, int tid) {
  int lane = tid & 31;
  int w    = tid >> 5;

  // Q/K/V projections: 48 16x16 tiles (head-dim GEMM, K padded 16->32)
  for (int t = w; t < 48; t += 8) {
    int which = t / 16;                // 0=Q (from Qsrc), 1=K, 2=V (from VKsrc)
    int tt = t & 15;
    int mt = tt >> 2, h = tt & 3;
    const float* sx = (which == 0) ? Qsrc : VKsrc;
    const _Float16* wm =
        Wt + (which == 0 ? OFF_WQ : (which == 1 ? OFF_WK : OFF_WV));
    float* dst = (which == 0) ? Qb : (which == 1 ? Kb : Vb);
    v16h a = loadA32_pad16(sx, mt * 16, h * 16, lane);
    v16h b = loadBpack(wm, lane);
    v8f  c = {};
    c = wmma_f16(a, b, c);
    storeC(dst, mt * 16, h * 16, lane, c);
  }
  __syncthreads();

  // per-head: scores -> softmax -> PV (PV output overwrites Q_h columns)
  for (int h = 0; h < NHEADS; ++h) {
    for (int t = w; t < 16; t += 8) {                  // S = Q_h K_h^T / 8
      int mt = t >> 2, nt = t & 3;
      v16h a = loadA32_pad16(Qb, mt * 16, h * 16, lane);
      v16h b = loadB_kT(Kb, nt * 16, h * 16, lane);
      v8f  c = {};
      c = wmma_f16(a, b, c);
#pragma unroll
      for (int r = 0; r < 8; ++r) c[r] *= 0.125f;      // 1/sqrt(64)
      storeC(Sb, mt * 16, nt * 16, lane, c);
    }
    __syncthreads();
    if (tid < TOK) {                                   // row softmax
      float* row = Sb + tid * XP;
      float mx = row[0];
#pragma unroll 8
      for (int c = 1; c < TOK; ++c) mx = fmaxf(mx, row[c]);
      float s = 0.f;
#pragma unroll 8
      for (int c = 0; c < TOK; ++c) { float e = __expf(row[c] - mx); row[c] = e; s += e; }
      float inv = 1.f / s;
#pragma unroll 8
      for (int c = 0; c < TOK; ++c) row[c] *= inv;
    }
    __syncthreads();
    if (w < 4) {                                       // out_h = P @ V_h
      int mt = w;
      v8f c = {};
#pragma unroll
      for (int kc = 0; kc < 2; ++kc) {
        v16h a = loadA32(Sb, mt * 16, kc * 32, lane);
        v16h b = loadB_V(Vb, kc * 32, h * 16, lane);
        c = wmma_f16(a, b, c);
      }
      storeC(Qb, mt * 16, h * 16, lane, c);            // Q_h dead -> reuse
    }
    __syncthreads();
  }

  // output projection: Sb = Qb @ fcw + fcb
  for (int t = w; t < 16; t += 8) {
    int mt = t >> 2, nt = t & 3;
    v8f c = {};
#pragma unroll
    for (int kc = 0; kc < 2; ++kc) {
      v16h a = loadA32(Qb, mt * 16, kc * 32, lane);
      v16h b = loadBpack(Wt + OFF_FCW + (kc * 4 + nt) * TILE_H, lane);
      c = wmma_f16(a, b, c);
    }
    float bias = fcb[nt * 16 + (lane & 15)];
#pragma unroll
    for (int r = 0; r < 8; ++r) c[r] += bias;
    storeC(Sb, mt * 16, nt * 16, lane, c);
  }
  __syncthreads();

  ln_resid(Xdst, Sb, Qsrc, lnw, lnb, tid);
  __syncthreads();
}

// ---------------------------------------------------------------------------
// FFN + residual + layernorm:  Xio = LN( relu(Xio@w1+b1)@w2 + b2 + Xio )
// ---------------------------------------------------------------------------
__device__ void ffn_ln(float* Xio, const _Float16* Wt,
                       const float* b1, const float* b2,
                       const float* lnw, const float* lnb,
                       _Float16* Hs, float* Sb, int tid) {
  int lane = tid & 31;
  int w    = tid >> 5;

  for (int t = w; t < 64; t += 8) {                    // H = relu(X@w1+b1), f16
    int mt = t >> 4, nt = t & 15;
    v8f c = {};
#pragma unroll
    for (int kc = 0; kc < 2; ++kc) {
      v16h a = loadA32(Xio, mt * 16, kc * 32, lane);
      v16h b = loadBpack(Wt + OFF_W1 + (kc * 16 + nt) * TILE_H, lane);
      c = wmma_f16(a, b, c);
    }
    storeH(Hs, mt * 16, nt * 16, lane, c, b1);
  }
  __syncthreads();

  for (int t = w; t < 16; t += 8) {                    // F = H@w2 + b2
    int mt = t >> 2, nt = t & 3;
    v8f c = {};
#pragma unroll
    for (int kc = 0; kc < 8; ++kc) {
      v16h a = loadAh(Hs, mt * 16, kc * 32, lane);
      v16h b = loadBpack(Wt + OFF_W2 + (kc * 4 + nt) * TILE_H, lane);
      c = wmma_f16(a, b, c);
    }
    float bias = b2[nt * 16 + (lane & 15)];
#pragma unroll
    for (int r = 0; r < 8; ++r) c[r] += bias;
    storeC(Sb, mt * 16, nt * 16, lane, c);
  }
  __syncthreads();

  ln_resid(Xio, Sb, Xio, lnw, lnb, tid);
  __syncthreads();
}

// ---------------------------------------------------------------------------
// Kernel 1: fold LSM sampling + init-recon into one 16x16 matrix P
// P[k][b] = sum_s sampling_w[s][k] * init_bl_w[b][s]
// ---------------------------------------------------------------------------
__global__ void k_precompute_P(PtrTable pt, float* P) {
  int t = threadIdx.x;                 // 256 threads, one P entry each
  int k = t >> 4, b = t & 15;
  const float* sw = pt.p[IDX_SAMP];    // [128][16]
  const float* ib = pt.p[IDX_INITBL];  // [16][128]
  float acc = 0.f;
  for (int s = 0; s < 128; ++s) acc += sw[s * 16 + k] * ib[b * 128 + s];
  P[k * 16 + b] = acc;
}

// ---------------------------------------------------------------------------
// Kernel 2: pack all GEMM weights to f16 in the WMMA B-operand lane layout.
// One 32-lane wave per 32x16 tile. 966 tiles total.
// ---------------------------------------------------------------------------
__global__ void k_pack_weights(PtrTable pt, _Float16* dst) {
  int rem = blockIdx.x;
  int u = 0;
  for (;;) {
    int cnt = (u >= 6 && u < 12) ? 11 : 75;   // SA units have 11 tiles, TB units 75
    if (rem < cnt) break;
    rem -= cnt;
    ++u;
  }
  int nm, si[6], Ks[6], Ns[6], of[6], dstBase;
  if (u < 6) {                                // encoder transformer block
    int b = ENC_BASE(u);
    nm = 6;
    si[0] = b + 0;  si[1] = b + 1;  si[2] = b + 2;
    si[3] = b + 3;  si[4] = b + 9;  si[5] = b + 11;
    Ks[0] = Ks[1] = Ks[2] = 16; Ns[0] = Ns[1] = Ns[2] = 16;
    Ks[3] = 64;  Ns[3] = 64;
    Ks[4] = 64;  Ns[4] = 256;
    Ks[5] = 256; Ns[5] = 64;
    of[0] = OFF_WV; of[1] = OFF_WK; of[2] = OFF_WQ;
    of[3] = OFF_FCW; of[4] = OFF_W1; of[5] = OFF_W2;
    dstBase = ENC_TB_OFF(u);
  } else if (u < 12) {                        // decoder self-attention
    int d0 = u - 6; int b = DEC_BASE(d0);
    nm = 4;
    si[0] = b + 0; si[1] = b + 1; si[2] = b + 2; si[3] = b + 3;
    Ks[0] = Ks[1] = Ks[2] = 16; Ns[0] = Ns[1] = Ns[2] = 16;
    Ks[3] = 64; Ns[3] = 64;
    of[0] = OFF_WV; of[1] = OFF_WK; of[2] = OFF_WQ; of[3] = OFF_FCW;
    dstBase = DEC_SA_OFF(d0);
  } else {                                    // decoder inner transformer block
    int d0 = u - 12; int b = DEC_BASE(d0);
    nm = 6;
    si[0] = b + 7;  si[1] = b + 8;  si[2] = b + 9;
    si[3] = b + 10; si[4] = b + 16; si[5] = b + 18;
    Ks[0] = Ks[1] = Ks[2] = 16; Ns[0] = Ns[1] = Ns[2] = 16;
    Ks[3] = 64;  Ns[3] = 64;
    Ks[4] = 64;  Ns[4] = 256;
    Ks[5] = 256; Ns[5] = 64;
    of[0] = OFF_WV; of[1] = OFF_WK; of[2] = OFF_WQ;
    of[3] = OFF_FCW; of[4] = OFF_W1; of[5] = OFF_W2;
    dstBase = DEC_TB_OFF(d0);
  }
  int m = 0;
  for (; m < nm; ++m) {
    int tiles = ((Ks[m] + 31) / 32) * (Ns[m] / 16);
    if (rem < tiles) break;
    rem -= tiles;
  }
  int K = Ks[m], Nn = Ns[m];
  int NT = Nn >> 4;
  int k0 = (rem / NT) * 32;
  int n0 = (rem % NT) * 16;
  const float* src = pt.p[si[m]];
  _Float16* d = dst + dstBase + of[m] + rem * TILE_H;

  int l  = threadIdx.x;                // 32 lanes
  int nn = n0 + (l & 15);
  int kb = (l >> 4) * 8;
#pragma unroll
  for (int i = 0; i < 16; ++i) {
    int k = k0 + kb + (i < 8 ? i : i + 8);
    d[l * 16 + i] = (k < K) ? (_Float16)src[k * Nn + nn] : (_Float16)0.f;
  }
}

// ---------------------------------------------------------------------------
// Kernel 3: one workgroup (8 wave32) per batch element, fully fused network.
// ---------------------------------------------------------------------------
__global__ __launch_bounds__(256)
void k_transformer(PtrTable pt, const float* Ppre, const _Float16* Wpk, float* out) {
  __shared__ float    Xs[TOK * XP];    // current stream
  __shared__ float    Es[TOK * XP];    // encoder output (K/V source for decoder)
  __shared__ float    Qs[TOK * XP];    // Q / attention concat
  __shared__ float    Ks[TOK * XP];    // K
  __shared__ float    Vs[TOK * XP];    // V
  __shared__ float    Ss[TOK * XP];    // scores / fc & ffn outputs
  __shared__ _Float16 Hs[TOK * HP];    // FF hidden (f16)
  __shared__ float    Pl[256];         // folded LSM matrix

  int tid  = threadIdx.x;
  int n    = blockIdx.x;

  Pl[tid] = Ppre[tid];
  __syncthreads();

  // ---- LSM init-recon + enc_pos:  Xs = (src patches @ P) + enc_pos --------
  {
    const float* src  = pt.p[IDX_SRC] + (size_t)n * 4096;
    const float* epos = pt.p[IDX_EPOS];
    int p = tid;                        // one 16-sample patch per thread
    float xv[16];
#pragma unroll
    for (int k = 0; k < 16; ++k) xv[k] = src[p * 16 + k];
    int row = p >> 2, cb = (p & 3) * 16;
#pragma unroll
    for (int b = 0; b < 16; ++b) {
      float acc = 0.f;
#pragma unroll
      for (int k = 0; k < 16; ++k) acc += xv[k] * Pl[k * 16 + b];
      Xs[row * XP + cb + b] = acc + epos[row * 64 + cb + b];
    }
  }
  __syncthreads();

  // ---- encoder ------------------------------------------------------------
  for (int e = 0; e < NLAYER; ++e) {
    const float* const* L = pt.p + ENC_BASE(e);
    const _Float16* Wt = Wpk + ENC_TB_OFF(e);
    attn_ln(Xs, Xs, Xs, Wt, L[4], L[5], L[6], Qs, Ks, Vs, Ss, tid);
    ffn_ln(Xs, Wt, L[10], L[12], L[7], L[8], Hs, Ss, tid);
  }

  // save encoder output
  for (int i = tid; i < TOK * XP; i += 256) Es[i] = Xs[i];
  __syncthreads();

  // ---- decoder input: trg + dec_pos ---------------------------------------
  {
    const float* trg  = pt.p[IDX_TRG] + (size_t)n * 4096;
    const float* dpos = pt.p[IDX_DPOS];
    for (int i = tid; i < 4096; i += 256)
      Xs[(i >> 6) * XP + (i & 63)] = trg[i] + dpos[i];
  }
  __syncthreads();

  // ---- decoder ------------------------------------------------------------
  for (int d = 0; d < NLAYER; ++d) {
    const float* const* L = pt.p + DEC_BASE(d);
    // self-attention:  Xs = LN(fc(MHA(Xs)) + Xs)      [lnw/lnb]
    attn_ln(Xs, Xs, Xs, Wpk + DEC_SA_OFF(d), L[4], L[5], L[6],
            Qs, Ks, Vs, Ss, tid);
    // cross-attention (blk): Xs = LN1(fc(MHA(Es,Es,Xs)) + Xs)
    const _Float16* Wt = Wpk + DEC_TB_OFF(d);
    attn_ln(Xs, Es, Xs, Wt, L[11], L[12], L[13], Qs, Ks, Vs, Ss, tid);
    // feed-forward: Xs = LN2(FF(Xs) + Xs)
    ffn_ln(Xs, Wt, L[17], L[19], L[14], L[15], Hs, Ss, tid);
  }

  // ---- write output -------------------------------------------------------
  float* o = out + (size_t)n * 4096;
  for (int i = tid; i < 4096; i += 256)
    o[i] = Xs[(i >> 6) * XP + (i & 63)];
}

// ---------------------------------------------------------------------------
extern "C" void kernel_launch(void* const* d_in, const int* in_sizes, int n_in,
                              void* d_out, int out_size, void* d_ws, size_t ws_size,
                              hipStream_t stream) {
  (void)in_sizes; (void)out_size; (void)ws_size;
  PtrTable pt;
  for (int i = 0; i < N_INPUTS; ++i)
    pt.p[i] = (i < n_in) ? (const float*)d_in[i] : nullptr;

  float*    P   = (float*)d_ws;                       // 256 f32
  _Float16* Wpk = (_Float16*)((char*)d_ws + 1024);    // ~0.99 MB packed f16 weights

  k_precompute_P<<<1, 256, 0, stream>>>(pt, P);
  k_pack_weights<<<966, 32, 0, stream>>>(pt, Wpk);
  k_transformer<<<NBATCH, 256, 0, stream>>>(pt, P, Wpk, (float*)d_out);
}
